// InputBlock_44169443672195
// MI455X (gfx1250) — compile-verified, matrix operand order checked by reading, not run on hardware
//
#include <hip/hip_runtime.h>
#include <hip/hip_bf16.h>

typedef __attribute__((ext_vector_type(16))) _Float16 v16h;
typedef __attribute__((ext_vector_type(8)))  float    v8f;

#define NN 50000
#define EE 800000
#define GG 32
#define HH 128

__device__ __forceinline__ float gelu_f(float x) {
    return 0.5f * x * (1.0f + erff(x * 0.7071067811865476f));
}

// Element->K mapping inside a 32-wide K-chunk for 16-bit WMMA fragments.
// lane<16: e in [0,8) -> K=e ; e in [8,16) -> K=e+8   (K in {0..7, 16..23})
// lane>=16: +8                                        (K in {8..15, 24..31})
__device__ __forceinline__ int kelem(int lane, int e) {
    return (e & 7) + ((e >> 3) << 4) + ((lane >> 4) << 3);
}

__device__ __forceinline__ v8f wmma_f16(v16h a, v16h b, v8f c) {
    return __builtin_amdgcn_wmma_f32_16x16x32_f16(false, a, false, b, (short)0, c,
                                                  false, false);
}

// ---------------------------------------------------------------------------
// Convert f32 weights [K,128] into f16 WMMA-B fragments:
//   half index = ((kc*8 + nt)*32 + lane)*16 + e
//     -> W[(kc*32 + kelem(lane,e))*128 + nt*16 + lane%16]
// ---------------------------------------------------------------------------
#define WNODE_HALVES 32768  // K=256: 8 kc * 8 nt * 512
#define WEDGE_HALVES 24576  // K=192: 6 kc * 8 nt * 512
#define WEF_HALVES   32768  // K=256
#define PREP_TOTAL   (WNODE_HALVES + WEDGE_HALVES + WEF_HALVES)

__global__ void __launch_bounds__(256) prep_weights_kernel(
        const float* __restrict__ Wnode,
        const float* __restrict__ Wedge,
        const float* __restrict__ Wef,
        _Float16* __restrict__ wn,
        _Float16* __restrict__ we,
        _Float16* __restrict__ wf) {
    int t = blockIdx.x * blockDim.x + threadIdx.x;
    if (t >= PREP_TOTAL) return;
    const float* src;
    _Float16* dst;
    if (t < WNODE_HALVES)                       { src = Wnode; dst = wn; }
    else if (t < WNODE_HALVES + WEDGE_HALVES)   { t -= WNODE_HALVES; src = Wedge; dst = we; }
    else                                        { t -= WNODE_HALVES + WEDGE_HALVES; src = Wef; dst = wf; }
    int frag   = t >> 9;
    int within = t & 511;
    int lane   = within >> 4;
    int e      = within & 15;
    int kc = frag >> 3, nt = frag & 7;
    int k = kc * 32 + kelem(lane, e);
    int n = nt * 16 + (lane & 15);
    dst[t] = (_Float16)src[k * 128 + n];
}

// ---------------------------------------------------------------------------
// cs = gelu(concat(emb_chg[charge], emb_spin[spin]) @ W_cs + b_cs)   [32,128]
// ---------------------------------------------------------------------------
__global__ void __launch_bounds__(128) cs_kernel(
        const int* __restrict__ charge, const int* __restrict__ spin,
        const float* __restrict__ emb_chg, const float* __restrict__ emb_spin,
        const float* __restrict__ W_cs, const float* __restrict__ b_cs,
        float* __restrict__ cs_out) {
    int g = blockIdx.x;
    int c = threadIdx.x;  // 128 threads
    const float* ec = emb_chg  + (size_t)charge[g] * 128;
    const float* es = emb_spin + (size_t)spin[g]   * 128;
    float s = b_cs[c];
    for (int k = 0; k < 128; ++k) s += ec[k] * W_cs[k * 128 + c];
    for (int k = 0; k < 128; ++k) s += es[k] * W_cs[(128 + k) * 128 + c];
    cs_out[g * 128 + c] = gelu_f(s);
}

// ---------------------------------------------------------------------------
// Node pipeline: gather atom emb + direction MLP -> LN -> [16,256]@[256,128]
// WMMA -> gelu -> + cs[node_batch] -> f16 node table.
// Block = 64 threads = 2 waves, 1 wave per 16-node tile.
// ---------------------------------------------------------------------------
__global__ void __launch_bounds__(64) node_kernel(
        const int* __restrict__ atomic_numbers,
        const float* __restrict__ ndexp,
        const int* __restrict__ node_batch,
        const float* __restrict__ emb_atom,
        const float* __restrict__ W_nd, const float* __restrict__ b_nd,
        const float* __restrict__ ln_g, const float* __restrict__ ln_b,
        const _Float16* __restrict__ Wnode_h,
        const float* __restrict__ b_node,
        const float* __restrict__ cs,
        _Float16* __restrict__ node_h) {
    __shared__ float sWnd[16 * 128];   // direction-MLP weight, block-shared
    __shared__ float sLng[256], sLnb[256];
    __shared__ float sX[2][16 * 256];  // per-wave concat(atom_e, nd_e)
    __shared__ float sNd[2][16 * 16];  // per-wave direction-expansion tile
    __shared__ float sMu[2][16], sRs[2][16];
    __shared__ int   sBatch[2][16];

    const int tid  = threadIdx.x;
    const int lane = tid & 31;
    const int wid  = tid >> 5;

    for (int i = tid; i < 16 * 128; i += 64) sWnd[i] = W_nd[i];
    for (int i = tid; i < 256; i += 64) { sLng[i] = ln_g[i]; sLnb[i] = ln_b[i]; }

    const int tile = blockIdx.x * 2 + wid;
    const bool valid = (tile * 16) < NN;
    const int base = tile * 16;

    if (valid) {
        for (int i = lane; i < 256; i += 32) sNd[wid][i] = ndexp[(size_t)base * 16 + i];
        if (lane < 16) sBatch[wid][lane] = node_batch[base + lane];
        for (int m = 0; m < 16; ++m) {
            int an = atomic_numbers[base + m];
            float4 v = *(const float4*)(emb_atom + (size_t)an * 128 + lane * 4);
            *(float4*)(&sX[wid][m * 256 + lane * 4]) = v;
        }
    }
    __syncthreads();

    if (valid) {
        const int c0 = lane * 4;
        for (int m = 0; m < 16; ++m) {
            float a0 = 0.f, a1 = 0.f, a2 = 0.f, a3 = 0.f;
#pragma unroll
            for (int k = 0; k < 16; ++k) {
                float xv = sNd[wid][m * 16 + k];
                const float* wr = &sWnd[k * 128 + c0];
                a0 += xv * wr[0]; a1 += xv * wr[1];
                a2 += xv * wr[2]; a3 += xv * wr[3];
            }
            float* xo = &sX[wid][m * 256 + 128 + c0];
            xo[0] = gelu_f(a0 + b_nd[c0 + 0]);
            xo[1] = gelu_f(a1 + b_nd[c0 + 1]);
            xo[2] = gelu_f(a2 + b_nd[c0 + 2]);
            xo[3] = gelu_f(a3 + b_nd[c0 + 3]);
        }
    }
    __syncthreads();

    if (valid && lane < 16) {
        const float* xr = &sX[wid][lane * 256];
        float s = 0.f, s2 = 0.f;
        for (int k = 0; k < 256; ++k) { float v = xr[k]; s += v; s2 += v * v; }
        float mu  = s * (1.f / 256.f);
        float var = s2 * (1.f / 256.f) - mu * mu;
        sMu[wid][lane] = mu;
        sRs[wid][lane] = rsqrtf(var + 1e-5f);
    }
    __syncthreads();

    // 16x256 @ 256x128 via 8 K-chunks x 8 N-tiles of v_wmma_f32_16x16x32_f16.
    // B fragments staged 4 at a time: keeps 8 b128 loads in flight per clause
    // without blowing the VGPR budget (acc64 + bf32 + a8 ~= 110 live).
    v8f acc[8];
    v8f zero = {};
#pragma unroll
    for (int nt = 0; nt < 8; ++nt) acc[nt] = zero;

    const int row = lane & 15;
    const float mu = sMu[wid][row];
    const float rs = sRs[wid][row];

#pragma unroll
    for (int kc = 0; kc < 8; ++kc) {
        v16h a;
#pragma unroll
        for (int e = 0; e < 16; ++e) {
            int k = kc * 32 + kelem(lane, e);
            float xv = (sX[wid][row * 256 + k] - mu) * rs * sLng[k] + sLnb[k];
            a[e] = (_Float16)xv;
        }
#pragma unroll
        for (int h = 0; h < 2; ++h) {
            v16h bf[4];
#pragma unroll
            for (int j = 0; j < 4; ++j)
                bf[j] = *(const v16h*)(Wnode_h +
                        ((size_t)(kc * 8 + h * 4 + j) * 32 + lane) * 16);
#pragma unroll
            for (int j = 0; j < 4; ++j)
                acc[h * 4 + j] = wmma_f16(a, bf[j], acc[h * 4 + j]);
        }
    }

    if (valid) {
        const int mhi = (lane >> 4) << 3;
#pragma unroll
        for (int nt = 0; nt < 8; ++nt) {
            int col = nt * 16 + row;
            float bb = b_node[col];
#pragma unroll
            for (int r = 0; r < 8; ++r) {
                int M = r + mhi;
                float v = gelu_f(acc[nt][r] + bb) + cs[(size_t)sBatch[wid][M] * 128 + col];
                node_h[(size_t)(base + M) * 128 + col] = (_Float16)v;
            }
        }
    }
}

// ---------------------------------------------------------------------------
// Edge pipeline (hot): eattr = gelu([dist||dir]@W_edge+b)      (K=192)
//                      out   = gelu([node[src]||eattr]@W_ef+b) (K=256)
// Block = 256 threads = 8 waves, 1 wave per 16-edge tile.
// ---------------------------------------------------------------------------
__global__ void __launch_bounds__(256) edge_kernel(
        const float* __restrict__ dist,
        const float* __restrict__ dire,
        const int* __restrict__ nsrc,
        const _Float16* __restrict__ node_h,
        const _Float16* __restrict__ Wedge_h,
        const _Float16* __restrict__ Wef_h,
        const float* __restrict__ b_edge,
        const float* __restrict__ b_ef,
        float* __restrict__ out) {
    __shared__ _Float16 sE[8][16 * 128];  // per-wave eattr tile (f16), 32 KB total
    __shared__ float sBe[128], sBf[128];

    const int tid  = threadIdx.x;
    const int lane = tid & 31;
    const int wid  = tid >> 5;

    for (int i = tid; i < 128; i += 256) { sBe[i] = b_edge[i]; sBf[i] = b_ef[i]; }
    __syncthreads();

    const int tile = blockIdx.x * 8 + wid;  // E/16 = 50000 tiles exactly
    const int base = tile * 16;
    const int row  = lane & 15;
    const int mhi  = (lane >> 4) << 3;

    // Hoist the random node gather: resolve neighbor_src now and prefetch the
    // 256B node row (L2-resident f16 table) so GEMM1's ~48 WMMAs hide the
    // gather latency.  Lowered to global_prefetch_b8 on gfx1250.
    const int src = nsrc[base + row];
    const _Float16* nodeRow = node_h + (size_t)src * 128;
    __builtin_prefetch(nodeRow, 0, 3);
    __builtin_prefetch(nodeRow + 64, 0, 3);

    v8f acc[8];
    v8f zero = {};
#pragma unroll
    for (int nt = 0; nt < 8; ++nt) acc[nt] = zero;

    const float* distRow = dist + (size_t)(base + row) * 128;
    const float* dirRow  = dire + (size_t)(base + row) * 64;

    // GEMM1: K=192 (dist: chunks 0..3, dir: chunks 4..5)
#pragma unroll
    for (int kc = 0; kc < 6; ++kc) {
        v16h a;
#pragma unroll
        for (int ep = 0; ep < 8; ++ep) {
            int e  = ep * 2;
            int kk = kelem(lane, e);  // even -> (kk, kk+1) contiguous
            float2 v;
            if (kc < 4) v = *(const float2*)(distRow + kc * 32 + kk);
            else        v = *(const float2*)(dirRow + (kc - 4) * 32 + kk);
            a[e]     = (_Float16)v.x;
            a[e + 1] = (_Float16)v.y;
        }
#pragma unroll
        for (int h = 0; h < 2; ++h) {
            v16h bf[4];
#pragma unroll
            for (int j = 0; j < 4; ++j)
                bf[j] = *(const v16h*)(Wedge_h +
                        ((size_t)(kc * 8 + h * 4 + j) * 32 + lane) * 16);
#pragma unroll
            for (int j = 0; j < 4; ++j)
                acc[h * 4 + j] = wmma_f16(a, bf[j], acc[h * 4 + j]);
        }
    }

    // eattr epilogue -> f16 LDS (A-operand staging for GEMM2)
#pragma unroll
    for (int nt = 0; nt < 8; ++nt) {
        int col = nt * 16 + row;
        float bb = sBe[col];
#pragma unroll
        for (int r = 0; r < 8; ++r) {
            sE[wid][(r + mhi) * 128 + col] = (_Float16)gelu_f(acc[nt][r] + bb);
        }
    }
    __syncthreads();

    // GEMM2: K=256 (node gather: chunks 0..3, eattr from LDS: chunks 4..7)
#pragma unroll
    for (int nt = 0; nt < 8; ++nt) acc[nt] = zero;

#pragma unroll
    for (int kc = 0; kc < 8; ++kc) {
        v16h a;
#pragma unroll
        for (int ep = 0; ep < 8; ++ep) {
            int e  = ep * 2;
            int kk = kelem(lane, e);
            const _Float16* p = (kc < 4)
                ? (nodeRow + kc * 32 + kk)
                : (&sE[wid][row * 128 + (kc - 4) * 32 + kk]);
            a[e]     = p[0];
            a[e + 1] = p[1];
        }
#pragma unroll
        for (int h = 0; h < 2; ++h) {
            v16h bf[4];
#pragma unroll
            for (int j = 0; j < 4; ++j)
                bf[j] = *(const v16h*)(Wef_h +
                        ((size_t)(kc * 8 + h * 4 + j) * 32 + lane) * 16);
#pragma unroll
            for (int j = 0; j < 4; ++j)
                acc[h * 4 + j] = wmma_f16(a, bf[j], acc[h * 4 + j]);
        }
    }

#pragma unroll
    for (int nt = 0; nt < 8; ++nt) {
        int col = nt * 16 + row;
        float bb = sBf[col];
#pragma unroll
        for (int r = 0; r < 8; ++r) {
            out[(size_t)(base + r + mhi) * 128 + col] = gelu_f(acc[nt][r] + bb);
        }
    }
}

// ---------------------------------------------------------------------------
extern "C" void kernel_launch(void* const* d_in, const int* in_sizes, int n_in,
                              void* d_out, int out_size, void* d_ws, size_t ws_size,
                              hipStream_t stream) {
    const int*   atomic_numbers = (const int*)d_in[0];
    const float* ndexp    = (const float*)d_in[1];
    const float* dist     = (const float*)d_in[2];
    const float* dire     = (const float*)d_in[3];
    const int*   charge   = (const int*)d_in[4];
    const int*   spin     = (const int*)d_in[5];
    const int*   node_bat = (const int*)d_in[6];
    const int*   nsrc     = (const int*)d_in[7];
    const float* emb_atom = (const float*)d_in[8];
    const float* W_nd     = (const float*)d_in[9];
    const float* b_nd     = (const float*)d_in[10];
    const float* ln_g     = (const float*)d_in[11];
    const float* ln_b     = (const float*)d_in[12];
    const float* W_node   = (const float*)d_in[13];
    const float* b_node   = (const float*)d_in[14];
    const float* emb_chg  = (const float*)d_in[15];
    const float* emb_spin = (const float*)d_in[16];
    const float* W_cs     = (const float*)d_in[17];
    const float* b_cs     = (const float*)d_in[18];
    const float* W_edge   = (const float*)d_in[19];
    const float* b_edge   = (const float*)d_in[20];
    const float* W_ef     = (const float*)d_in[21];
    const float* b_ef     = (const float*)d_in[22];
    float* out = (float*)d_out;

    // Workspace layout (all offsets 256B-aligned)
    char* ws = (char*)d_ws;
    float*    cs       = (float*)(ws + 0);                       //   16 KB
    _Float16* node_h   = (_Float16*)(ws + 16384);                // 12.8 MB
    _Float16* Wnode_h  = (_Float16*)(ws + 12816384);             //   64 KB
    _Float16* Wedge_h  = (_Float16*)(ws + 12881920);             //   48 KB
    _Float16* Wef_h    = (_Float16*)(ws + 12931072);             //   64 KB

    prep_weights_kernel<<<(PREP_TOTAL + 255) / 256, 256, 0, stream>>>(
        W_node, W_edge, W_ef, Wnode_h, Wedge_h, Wef_h);

    cs_kernel<<<GG, 128, 0, stream>>>(charge, spin, emb_chg, emb_spin, W_cs, b_cs, cs);

    const int node_tiles = NN / 16;  // 3125
    node_kernel<<<(node_tiles + 1) / 2, 64, 0, stream>>>(
        atomic_numbers, ndexp, node_bat, emb_atom, W_nd, b_nd, ln_g, ln_b,
        Wnode_h, b_node, cs, node_h);

    const int edge_tiles = EE / 16;  // 50000
    edge_kernel<<<edge_tiles / 8, 256, 0, stream>>>(
        dist, dire, nsrc, node_h, Wedge_h, Wef_h, b_edge, b_ef, out);
}
